// SeparableConvGRU3D_20298015441340
// MI455X (gfx1250) — compile-verified, matrix operand order checked by reading, not run on hardware
//
#include <hip/hip_runtime.h>

// ---------------------------------------------------------------------------
// ConvGRU3D (B=1, D=32, H=64, W=64, C=32, K=3) as implicit GEMM on CDNA5 WMMA.
//   conv = [131072 x 864] * [864 x 32]; K-loop = 27 taps x (K=32 per WMMA).
//   Each wave owns an M=32 x N=32 output tile (two 16-pt groups along W) so
//   every LDS B-operand feeds two WMMAs (1 ds_load_b128 per wmma).
//   Tap loop = runtime loop over 9 (kd,kh) rows x unrolled kw(3): keeps all DS
//   offsets in 16-bit immediates off one per-iteration base -> no addr spills.
//   phase1 (gates): r = sig(Wr*x+Ur*h+br), z = sig(Wz*x+Uz*h+bz)
//                   -> 16 wmma/tap (8 accs); writes z (f32) and rh=(f16)(r*h).
//   phase2 (cand):  acc = Wc*x + Uc*(r*h); out = z*h + (1-z)*relu(acc + bh)
//                   -> 8 wmma/tap (4 accs).
// ---------------------------------------------------------------------------

typedef __attribute__((ext_vector_type(16))) _Float16 v16h;
typedef __attribute__((ext_vector_type(8)))  _Float16 v8h;
typedef __attribute__((ext_vector_type(8)))  float    v8f;

#define DD 32
#define HH 64
#define WW 64
#define CC 32
constexpr int PD = DD + 2, PH = HH + 2, PW = WW + 2;
constexpr int NPTS   = DD * HH * WW;            // 131072 points
constexpr int PPTS   = PD * PH * PW;            // padded points
constexpr int WMAT   = 27 * 2 * 32 * 16;        // halves per repacked matrix
constexpr int TILE32 = NPTS / 32;               // 4096 M=32 wave tiles

// ---------------- prep kernels ----------------

__global__ void pad_cvt(const float* __restrict__ src, _Float16* __restrict__ dst) {
  int total = PPTS * CC;
  for (int i = blockIdx.x * blockDim.x + threadIdx.x; i < total;
       i += gridDim.x * blockDim.x) {
    int c = i & (CC - 1);
    int p = i >> 5;
    int px = p % PW; int t = p / PW;
    int py = t % PH; int pd = t / PH;
    int d = pd - 1, y = py - 1, x = px - 1;
    float v = 0.0f;
    if ((unsigned)d < (unsigned)DD && (unsigned)y < (unsigned)HH &&
        (unsigned)x < (unsigned)WW)
      v = src[((d * HH + y) * WW + x) * CC + c];
    dst[i] = (_Float16)v;
  }
}

// Repack [kd][kh][kw][cin][cout] f32 -> per-lane WMMA-B layout, interleaved by
// tap across nmats matrices: idx = ((tap*nmats + mat)*2 + nt)*512 + lane*16 + j.
__global__ void repack_w(const float* __restrict__ src, _Float16* __restrict__ dst,
                         int mat, int nmats) {
  int i = blockIdx.x * blockDim.x + threadIdx.x;
  if (i >= WMAT) return;
  int j     = i & 15;
  int lane  = (i >> 4) & 31;
  int ntile = (i >> 9) & 1;
  int tap   = i >> 10;
  int n = ntile * 16 + (lane & 15);
  int k = ((lane >> 4) << 3) + (j < 8 ? j : j + 8);
  dst[((tap * nmats + mat) * 2 + ntile) * 512 + lane * 16 + j] =
      (_Float16)src[tap * (CC * CC) + k * CC + n];
}

// ---------------- helpers ----------------

__device__ __forceinline__ v16h loadA(const _Float16* p) {
  union { v16h v; v8h h[2]; } a;
  a.h[0] = *(const v8h*)(p);
  a.h[1] = *(const v8h*)(p + 16);
  return a.v;
}

__device__ __forceinline__ float fsig(float t) { return 1.0f / (1.0f + __expf(-t)); }

#define WMMA16(A, B, Cacc) \
  __builtin_amdgcn_wmma_f32_16x16x32_f16(false, (A), false, (B), (short)0, (Cacc), false, false)

__device__ __forceinline__ void gates_epi(const v8f& ar0, const v8f& ar1,
                                          const v8f& az0, const v8f& az1,
                                          int pbase, int prbase, int mbase, int n,
                                          float rb0, float rb1, float zb0, float zb1,
                                          const float* __restrict__ hsrc,
                                          float* __restrict__ z_out,
                                          _Float16* __restrict__ rh_out) {
#pragma unroll
  for (int v = 0; v < 8; ++v) {
    int idx  = (pbase + mbase + v) * CC + n;
    int pidx = (prbase + mbase + v) * CC + n;
    float h0 = hsrc[idx], h1 = hsrc[idx + 16];
    float r0 = fsig(ar0[v] + rb0);
    float r1 = fsig(ar1[v] + rb1);
    z_out[idx]        = fsig(az0[v] + zb0);
    z_out[idx + 16]   = fsig(az1[v] + zb1);
    rh_out[pidx]      = (_Float16)(r0 * h0);
    rh_out[pidx + 16] = (_Float16)(r1 * h1);
  }
}

__device__ __forceinline__ void cand_epi(const v8f& c0, const v8f& c1,
                                         int pbase, int mbase, int n,
                                         float bh0, float bh1,
                                         const float* __restrict__ z,
                                         const float* __restrict__ hsrc,
                                         float* __restrict__ out) {
#pragma unroll
  for (int v = 0; v < 8; ++v) {
    int idx = (pbase + mbase + v) * CC + n;
    float t0 = c0[v] + bh0; t0 = t0 > 0.0f ? t0 : 0.0f;
    float t1 = c1[v] + bh1; t1 = t1 > 0.0f ? t1 : 0.0f;
    float z0 = z[idx],      h0 = hsrc[idx];
    float z1 = z[idx + 16], h1 = hsrc[idx + 16];
    out[idx]      = z0 * h0 + (1.0f - z0) * t0;
    out[idx + 16] = z1 * h1 + (1.0f - z1) * t1;
  }
}

// ---------------- phase 1: r/z gates, fused r*h production ----------------

__global__ void __launch_bounds__(512)
gates_wmma(const _Float16* __restrict__ xpad, const _Float16* __restrict__ hpad,
           const _Float16* __restrict__ wB4,  // [tap][Wr,Ur,Wz,Uz][nt][lane][j]
           const float* __restrict__ br, const float* __restrict__ bz,
           const float* __restrict__ hsrc,
           float* __restrict__ z_out, _Float16* __restrict__ rh_out) {
  extern __shared__ _Float16 sW[];            // 4*WMAT halves = 221184 B
  const int NSTG = 4 * WMAT / 8;
  for (int i = threadIdx.x; i < NSTG; i += blockDim.x)
    ((v8h*)sW)[i] = ((const v8h*)wB4)[i];
  __syncthreads();

  int lane = threadIdx.x & 31;
  int wave = (blockIdx.x * blockDim.x + threadIdx.x) >> 5;
  if (wave >= TILE32) return;
  int w0 = (wave & 1) << 5;                   // 0 or 32
  int hy = (wave >> 1) & 63;
  int dz = wave >> 7;
  int m = lane & 15;
  int chalf = (lane >> 4) << 3;

  const long abase = ((long)(dz * PH + hy) * PW + (w0 + m)) * CC + chalf;
  const _Float16* pAx = xpad + abase;
  const _Float16* pAh = hpad + abase;
  const _Float16* sB = sW + lane * 16;

  v8f arA0 = {}, arA1 = {}, azA0 = {}, azA1 = {};   // tile A (w0..w0+15)
  v8f arB0 = {}, arB1 = {}, azB0 = {}, azB1 = {};   // tile B (w0+16..w0+31)

#pragma unroll 1
  for (int it = 0; it < 9; ++it) {            // it = kd*3 + kh
    const int kd = it / 3;
    const int kh = it - kd * 3;
    const int arow = (kd * PH + kh) * PW * CC;      // runtime row offset
    const _Float16* bt = sB + it * 12288;           // 3 taps * 4096 halves
#pragma unroll
    for (int kw = 0; kw < 3; ++kw) {
      const int aoff = arow + kw * CC;              // + small immediates
      v16h axA = loadA(pAx + aoff);
      v16h axB = loadA(pAx + aoff + 16 * CC);
      v16h ahA = loadA(pAh + aoff);
      v16h ahB = loadA(pAh + aoff + 16 * CC);
      const _Float16* bk = bt + kw * 4096;          // all DS offsets < 24KB
      v16h bwr0 = *(const v16h*)(bk + 0 * 512);
      v16h bwr1 = *(const v16h*)(bk + 1 * 512);
      v16h bur0 = *(const v16h*)(bk + 2 * 512);
      v16h bur1 = *(const v16h*)(bk + 3 * 512);
      v16h bwz0 = *(const v16h*)(bk + 4 * 512);
      v16h bwz1 = *(const v16h*)(bk + 5 * 512);
      v16h buz0 = *(const v16h*)(bk + 6 * 512);
      v16h buz1 = *(const v16h*)(bk + 7 * 512);
      // each B operand feeds two WMMAs; same-D pairs are 8 apart
      arA0 = WMMA16(axA, bwr0, arA0);  arB0 = WMMA16(axB, bwr0, arB0);
      azA0 = WMMA16(axA, bwz0, azA0);  azB0 = WMMA16(axB, bwz0, azB0);
      arA1 = WMMA16(axA, bwr1, arA1);  arB1 = WMMA16(axB, bwr1, arB1);
      azA1 = WMMA16(axA, bwz1, azA1);  azB1 = WMMA16(axB, bwz1, azB1);
      arA0 = WMMA16(ahA, bur0, arA0);  arB0 = WMMA16(ahB, bur0, arB0);
      azA0 = WMMA16(ahA, buz0, azA0);  azB0 = WMMA16(ahB, buz0, azB0);
      arA1 = WMMA16(ahA, bur1, arA1);  arB1 = WMMA16(ahB, bur1, arB1);
      azA1 = WMMA16(ahA, buz1, azA1);  azB1 = WMMA16(ahB, buz1, azB1);
    }
  }

  int n = lane & 15;
  int mbase = (lane >> 4) << 3;
  int pbase  = (dz * HH + hy) * WW + w0;
  int prbase = ((dz + 1) * PH + (hy + 1)) * PW + (w0 + 1);
  float rb0 = br[n], rb1 = br[n + 16];
  float zb0 = bz[n], zb1 = bz[n + 16];
  gates_epi(arA0, arA1, azA0, azA1, pbase,      prbase,      mbase, n,
            rb0, rb1, zb0, zb1, hsrc, z_out, rh_out);
  gates_epi(arB0, arB1, azB0, azB1, pbase + 16, prbase + 16, mbase, n,
            rb0, rb1, zb0, zb1, hsrc, z_out, rh_out);
}

// ---------------- phase 2: Wc*x + Uc*(r*h) + fused GRU update ----------------

__global__ void __launch_bounds__(512)
cand_wmma(const _Float16* __restrict__ xpad, const _Float16* __restrict__ rhpad,
          const _Float16* __restrict__ wB2,  // [tap][Wc,Uc][nt][lane][j]
          const float* __restrict__ z, const float* __restrict__ hsrc,
          const float* __restrict__ bh, float* __restrict__ out) {
  extern __shared__ _Float16 sU[];            // 2*WMAT halves = 110592 B
  const int NSTG = 2 * WMAT / 8;
  for (int i = threadIdx.x; i < NSTG; i += blockDim.x)
    ((v8h*)sU)[i] = ((const v8h*)wB2)[i];
  __syncthreads();

  int lane = threadIdx.x & 31;
  int wave = (blockIdx.x * blockDim.x + threadIdx.x) >> 5;
  if (wave >= TILE32) return;
  int w0 = (wave & 1) << 5;
  int hy = (wave >> 1) & 63;
  int dz = wave >> 7;
  int m = lane & 15;
  int chalf = (lane >> 4) << 3;

  const long abase = ((long)(dz * PH + hy) * PW + (w0 + m)) * CC + chalf;
  const _Float16* pAx = xpad + abase;
  const _Float16* pAr = rhpad + abase;
  const _Float16* sB = sU + lane * 16;

  v8f cA0 = {}, cA1 = {}, cB0 = {}, cB1 = {};

#pragma unroll 1
  for (int it = 0; it < 9; ++it) {            // it = kd*3 + kh
    const int kd = it / 3;
    const int kh = it - kd * 3;
    const int arow = (kd * PH + kh) * PW * CC;
    const _Float16* bt = sB + it * 6144;            // 3 taps * 2048 halves
#pragma unroll
    for (int kw = 0; kw < 3; ++kw) {
      const int aoff = arow + kw * CC;
      v16h axA = loadA(pAx + aoff);
      v16h axB = loadA(pAx + aoff + 16 * CC);
      v16h arA = loadA(pAr + aoff);
      v16h arB = loadA(pAr + aoff + 16 * CC);
      const _Float16* bk = bt + kw * 2048;          // all DS offsets < 12KB
      v16h bwc0 = *(const v16h*)(bk + 0 * 512);
      v16h bwc1 = *(const v16h*)(bk + 1 * 512);
      v16h buc0 = *(const v16h*)(bk + 2 * 512);
      v16h buc1 = *(const v16h*)(bk + 3 * 512);
      cA0 = WMMA16(axA, bwc0, cA0);  cB0 = WMMA16(axB, bwc0, cB0);
      cA1 = WMMA16(axA, bwc1, cA1);  cB1 = WMMA16(axB, bwc1, cB1);
      cA0 = WMMA16(arA, buc0, cA0);  cB0 = WMMA16(arB, buc0, cB0);
      cA1 = WMMA16(arA, buc1, cA1);  cB1 = WMMA16(arB, buc1, cB1);
    }
  }

  int n = lane & 15;
  int mbase = (lane >> 4) << 3;
  int pbase = (dz * HH + hy) * WW + w0;
  float bh0 = bh[n], bh1 = bh[n + 16];
  cand_epi(cA0, cA1, pbase,      mbase, n, bh0, bh1, z, hsrc, out);
  cand_epi(cB0, cB1, pbase + 16, mbase, n, bh0, bh1, z, hsrc, out);
}

// ---------------- host launch ----------------

extern "C" void kernel_launch(void* const* d_in, const int* in_sizes, int n_in,
                              void* d_out, int out_size, void* d_ws, size_t ws_size,
                              hipStream_t stream) {
  (void)in_sizes; (void)n_in; (void)out_size; (void)ws_size;
  const float* x  = (const float*)d_in[0];
  const float* h  = (const float*)d_in[1];
  const float* Wr = (const float*)d_in[2];
  const float* Ur = (const float*)d_in[3];
  const float* Wz = (const float*)d_in[4];
  const float* Uz = (const float*)d_in[5];
  const float* Wc = (const float*)d_in[6];
  const float* Uc = (const float*)d_in[7];
  const float* br = (const float*)d_in[8];
  const float* bz = (const float*)d_in[9];
  const float* bh = (const float*)d_in[10];

  const size_t NEL  = (size_t)NPTS * CC;
  const size_t PNEL = (size_t)PPTS * CC;
  char* ws = (char*)d_ws;
  size_t off = 0;
  _Float16* xpad  = (_Float16*)(ws + off); off += PNEL * 2;
  _Float16* hpad  = (_Float16*)(ws + off); off += PNEL * 2;
  _Float16* rhpad = (_Float16*)(ws + off); off += PNEL * 2;
  _Float16* wB4   = (_Float16*)(ws + off); off += (size_t)4 * WMAT * 2;
  _Float16* wB2   = (_Float16*)(ws + off); off += (size_t)2 * WMAT * 2;
  float* zbuf = (float*)(ws + off); off += NEL * 4;

  pad_cvt<<<1024, 256, 0, stream>>>(x, xpad);
  pad_cvt<<<1024, 256, 0, stream>>>(h, hpad);
  hipMemsetAsync(rhpad, 0, PNEL * 2, stream);   // zero halo (interior overwritten)
  int rblk = (WMAT + 255) / 256;
  repack_w<<<rblk, 256, 0, stream>>>(Wr, wB4, 0, 4);
  repack_w<<<rblk, 256, 0, stream>>>(Ur, wB4, 1, 4);
  repack_w<<<rblk, 256, 0, stream>>>(Wz, wB4, 2, 4);
  repack_w<<<rblk, 256, 0, stream>>>(Uz, wB4, 3, 4);
  repack_w<<<rblk, 256, 0, stream>>>(Wc, wB2, 0, 2);
  repack_w<<<rblk, 256, 0, stream>>>(Uc, wB2, 1, 2);

  gates_wmma<<<TILE32 / 16, 512, 4 * WMAT * 2, stream>>>(xpad, hpad, wB4, br, bz,
                                                         h, zbuf, rhpad);
  cand_wmma<<<TILE32 / 16, 512, 2 * WMAT * 2, stream>>>(xpad, rhpad, wB2, zbuf, h,
                                                        bh, (float*)d_out);
}
